// AFNONet_6176162971875
// MI455X (gfx1250) — matrix operand order repaired and checked by reading, not compiled, not run on hardware
//
#include <hip/hip_runtime.h>
#include <hip/hip_bf16.h>

// ---------------------------------------------------------------------------
// AFNONet forward for gfx1250 (MI455X). All heavy GEMMs use
// v_wmma_f32_16x16x32_bf16 (bf16 inputs, fp32 accumulate). Wave32 layouts per
// CDNA5 ISA 7.12.2. LDS tiles are stored in lane-fragment order so each WMMA
// operand loads as two aligned ds_load_b128 (no register shuffling).
// ---------------------------------------------------------------------------

typedef __attribute__((ext_vector_type(16))) __bf16    v16bf;
typedef __attribute__((ext_vector_type(8)))  float     v8f;
typedef __attribute__((ext_vector_type(4)))  unsigned  v4u;
typedef __attribute__((ext_vector_type(2)))  unsigned  v2u;

union Frag16 { v4u q[2]; v16bf v; };

__device__ __forceinline__ unsigned f2bf1(float x) {           // fp32 -> bf16 RNE
  unsigned u = __float_as_uint(x);
  return (u + 0x7FFFu + ((u >> 16) & 1u)) >> 16;
}
__device__ __forceinline__ unsigned packbf(float lo, float hi) {
  return f2bf1(lo) | (f2bf1(hi) << 16);
}
__device__ __forceinline__ v8f vzero() {
  v8f z;
#pragma unroll
  for (int i = 0; i < 8; ++i) z[i] = 0.f;
  return z;
}
// Fragment load: 8 contiguous u32 (16B-aligned base) -> two ds_load_b128.
__device__ __forceinline__ v16bf frag8(const unsigned* lds, int base) {
  Frag16 f;
  f.q[0] = *(const v4u*)(lds + base);
  f.q[1] = *(const v4u*)(lds + base + 4);
  return f.v;
}
__device__ __forceinline__ v16bf negbf(v16bf a) {  // negate 16 bf16 lanes
  Frag16 f; f.v = a;
#pragma unroll
  for (int i = 0; i < 2; ++i) f.q[i] ^= (v4u)(0x80008000u);
  return f.v;
}
// A-layout swizzle: kpair kp (0..15 within a 32-K tile) -> word slot
//   half = (kp>>2)&1 ; slot = (kp&3) + ((kp>>3)<<2) ; addr = half*8 + slot
__device__ __forceinline__ int aswz(int kp) {
  return (((kp >> 2) & 1) << 3) + (kp & 3) + ((kp >> 3) << 2);
}
#define WMMA_BF16(A, B, C) \
  __builtin_amdgcn_wmma_f32_16x16x32_bf16(false, (A), false, (B), (short)0, (C), false, false)

// ---------------------------------------------------------------------------
// Generic tiled GEMM: Out(MxN) = act(A(MxK) @ Bw(NxK)^T + bias) [+extra]
// EPI: 0=bias  1=bias+pos_embed[m%400]  2=bias+GELU(exact)
//      3=bias+residual extra[m][n]      4=bias+head pixel scatter
// ---------------------------------------------------------------------------
template <int EPI>
__global__ __launch_bounds__(256) void gemm_bf16(
    const float* __restrict__ A, const float* __restrict__ Bw,
    const float* __restrict__ bias, const float* __restrict__ extra,
    float* __restrict__ Out, int M, int N, int K) {
  __shared__ unsigned ldsA[128 * 20];  // [m][frag-order 16 kpairs], pitch 20
  __shared__ unsigned ldsB[128 * 20];  // [n][frag-order 16 kpairs], pitch 20

  const int tid = threadIdx.x;
  const int wave = tid >> 5, lane = tid & 31, nl = lane & 15, half = lane >> 4;
  const int wm = wave & 1, wn = wave >> 1;         // 2x4 wave grid -> 128x128
  const int m0 = blockIdx.y * 128, n0 = blockIdx.x * 128;
  const int sm = tid & 127, ks = tid >> 7;

  v8f acc[4][2];
#pragma unroll
  for (int mi = 0; mi < 4; ++mi)
#pragma unroll
    for (int ni = 0; ni < 2; ++ni) acc[mi][ni] = vzero();

  for (int k0 = 0; k0 < K; k0 += 32) {
    __syncthreads();
    {  // stage A tile 128x32 fp32 -> packed bf16, fragment-order swizzle
      const float* ap = A + (size_t)(m0 + sm) * K + k0;
      if (k0 + 32 < K) __builtin_prefetch(ap + 32, 0, 1);
#pragma unroll
      for (int j = 0; j < 4; ++j) {
        int q = ks * 4 + j;                        // kpairs 2q, 2q+1
        float4 v = *(const float4*)(ap + 4 * q);
        v2u w; w.x = packbf(v.x, v.y); w.y = packbf(v.z, v.w);
        *(v2u*)(ldsA + sm * 20 + aswz(2 * q)) = w; // 2q,2q+1 adjacent slots
      }
    }
    {  // stage B tile 32x128: weights are [n][k] row-major (k contiguous)
      int gn = n0 + sm;
      if (gn < N) {
        const float* bp = Bw + (size_t)gn * K + k0;
#pragma unroll
        for (int j = 0; j < 4; ++j) {
          int q = ks * 4 + j;
          float4 v = *(const float4*)(bp + 4 * q);
          v2u w; w.x = packbf(v.x, v.y); w.y = packbf(v.z, v.w);
          *(v2u*)(ldsB + sm * 20 + aswz(2 * q)) = w;
        }
      } else {
        v2u w; w.x = 0u; w.y = 0u;
#pragma unroll
        for (int j = 0; j < 4; ++j) *(v2u*)(ldsB + sm * 20 + 2 * (ks * 4 + j)) = w;
      }
    }
    __syncthreads();

    v16bf af[4], bf2[2];
#pragma unroll
    for (int mi = 0; mi < 4; ++mi)
      af[mi] = frag8(ldsA, (wm * 64 + mi * 16 + nl) * 20 + half * 8);
#pragma unroll
    for (int ni = 0; ni < 2; ++ni)
      bf2[ni] = frag8(ldsB, (wn * 32 + ni * 16 + nl) * 20 + half * 8);
#pragma unroll
    for (int mi = 0; mi < 4; ++mi)
#pragma unroll
      for (int ni = 0; ni < 2; ++ni)
        acc[mi][ni] = WMMA_BF16(af[mi], bf2[ni], acc[mi][ni]);
  }

  // epilogue. C/D layout: VGPR r, lanes(half) -> m = r + 8*half, n = nl.
#pragma unroll
  for (int mi = 0; mi < 4; ++mi) {
#pragma unroll
    for (int ni = 0; ni < 2; ++ni) {
      int gn = n0 + wn * 32 + ni * 16 + nl;
      if (gn >= N) continue;
      float bv = bias[gn];
#pragma unroll
      for (int r = 0; r < 8; ++r) {
        int gm = m0 + wm * 64 + mi * 16 + half * 8 + r;
        float v = acc[mi][ni][r] + bv;
        if constexpr (EPI == 1) v += extra[(size_t)(gm % 400) * N + gn];
        if constexpr (EPI == 2) v = 0.5f * v * (1.f + erff(v * 0.70710678f));
        if constexpr (EPI == 3) v += extra[(size_t)gm * N + gn];
        if constexpr (EPI == 4) {
          int b = gm / 400, tok = gm % 400, gh = tok / 20, gw = tok % 20;
          int i = gn / 552, rem = gn % 552, j = rem / 69, oc = rem % 69;
          Out[((size_t)(b * 69 + oc) * 160 + (gh * 8 + i)) * 160 + (gw * 8 + j)] = v;
        } else {
          Out[(size_t)gm * N + gn] = v;
        }
      }
    }
  }
}

// ---------------------------------------------------------------------------
// Fused AFNO frequency MLP: per (nb, 64-position tile)
//   o1 = relu(x @ W1 + b1)  (complex, 96x96 per block)
//   o2 = o1 @ W2 + b2 ; soft-shrink by radial threshold
// K=96 => 48 kpairs; A-tiles use pitch 52 = 3 k-tiles * 16 + 4 pad.
// ---------------------------------------------------------------------------
__global__ __launch_bounds__(256) void afno_kernel(
    const float* __restrict__ Fr, const float* __restrict__ Fi,
    const float* __restrict__ w1, const float* __restrict__ b1,
    const float* __restrict__ w2, const float* __restrict__ b2,
    float* __restrict__ Zr, float* __restrict__ Zi) {
  __shared__ unsigned wW[4][96 * 52];   // W1r,W1i,W2r,W2i packed bf16 [o][kp]
  __shared__ unsigned xP[2][64 * 52];   // xr,xi A-swizzled [m][ktile|half|slot]
  __shared__ unsigned o1P[2][64 * 52];  // layer-1 output, same layout
  __shared__ float    bia[4][96];

  const int tid = threadIdx.x, nb = blockIdx.y, pt = blockIdx.x;
  const float* wsrc[4] = {w1 + (size_t)nb * 9216, w1 + (size_t)(8 + nb) * 9216,
                          w2 + (size_t)nb * 9216, w2 + (size_t)(8 + nb) * 9216};
  const float* bsrc[4] = {b1 + nb * 96, b1 + (8 + nb) * 96,
                          b2 + nb * 96, b2 + (8 + nb) * 96};
  for (int mat = 0; mat < 4; ++mat)
    for (int idx = tid; idx < 48 * 96; idx += 256) {
      int kp = idx / 96, o = idx % 96;   // weights [i][o]: pair (2kp,2kp+1) in i
      wW[mat][o * 52 + (kp >> 4) * 16 + aswz(kp & 15)] = packbf(wsrc[mat][(2 * kp) * 96 + o],
                                    wsrc[mat][(2 * kp + 1) * 96 + o]);
    }
  if (tid < 96)
    for (int m2 = 0; m2 < 4; ++m2) bia[m2][tid] = bsrc[m2][tid];
  for (int idx = tid; idx < 64 * 24; idx += 256) {  // float4 = kpairs 2q,2q+1
    int m = idx / 24, q = idx % 24;
    size_t g = (size_t)(pt * 64 + m) * 768 + nb * 96 + 4 * q;
    int kp = 2 * q, kt = kp >> 4, kpl = kp & 15;
    int a32 = m * 52 + kt * 16 + aswz(kpl);
    float4 vr = *(const float4*)(Fr + g);
    float4 vi = *(const float4*)(Fi + g);
    v2u wr; wr.x = packbf(vr.x, vr.y); wr.y = packbf(vr.z, vr.w);
    v2u wi; wi.x = packbf(vi.x, vi.y); wi.y = packbf(vi.z, vi.w);
    *(v2u*)(xP[0] + a32) = wr;
    *(v2u*)(xP[1] + a32) = wi;
  }
  __syncthreads();

  const int wave = tid >> 5, lane = tid & 31, nl = lane & 15, half = lane >> 4;
  const int mt = wave >> 1, nh = wave & 1;  // wave: 16 rows x 48 cols

  // ---- layer 1: cr = xr*Wr - xi*Wi ; ci = xi*Wr + xr*Wi
  v8f cr[3], ci[3];
#pragma unroll
  for (int j = 0; j < 3; ++j) { cr[j] = vzero(); ci[j] = vzero(); }
#pragma unroll
  for (int kt = 0; kt < 3; ++kt) {
    int abase = (mt * 16 + nl) * 52 + kt * 16 + half * 8;
    v16bf ar = frag8(xP[0], abase);
    v16bf ai = frag8(xP[1], abase);
    v16bf nai = negbf(ai);  // bf16 WMMA has no A-negate; flip sign bits
#pragma unroll
    for (int j = 0; j < 3; ++j) {
      int col = nh * 48 + j * 16 + nl;
      v16bf br = frag8(wW[0], col * 52 + kt * 16 + half * 8);
      v16bf bi = frag8(wW[1], col * 52 + kt * 16 + half * 8);
      cr[j] = WMMA_BF16(ar, br, cr[j]);
      cr[j] = WMMA_BF16(nai, bi, cr[j]);
      ci[j] = WMMA_BF16(ai, br, ci[j]);
      ci[j] = WMMA_BF16(ar, bi, ci[j]);
    }
  }
  unsigned short* o1rh = (unsigned short*)o1P[0];
  unsigned short* o1ih = (unsigned short*)o1P[1];
#pragma unroll
  for (int j = 0; j < 3; ++j) {
    int n = nh * 48 + j * 16 + nl;
    int kpn = n >> 1, kt = kpn >> 4, kpl = kpn & 15;
    float brv = bia[0][n], biv = bia[1][n];
#pragma unroll
    for (int r = 0; r < 8; ++r) {
      int m = mt * 16 + half * 8 + r;
      int a32 = m * 52 + kt * 16 + aswz(kpl);
      o1rh[a32 * 2 + (n & 1)] = (unsigned short)f2bf1(fmaxf(cr[j][r] + brv, 0.f));
      o1ih[a32 * 2 + (n & 1)] = (unsigned short)f2bf1(fmaxf(ci[j][r] + biv, 0.f));
    }
  }
  __syncthreads();

  // ---- layer 2
  v8f dr[3], di[3];
#pragma unroll
  for (int j = 0; j < 3; ++j) { dr[j] = vzero(); di[j] = vzero(); }
#pragma unroll
  for (int kt = 0; kt < 3; ++kt) {
    int abase = (mt * 16 + nl) * 52 + kt * 16 + half * 8;
    v16bf ar = frag8(o1P[0], abase);
    v16bf ai = frag8(o1P[1], abase);
    v16bf nai = negbf(ai);
#pragma unroll
    for (int j = 0; j < 3; ++j) {
      int col = nh * 48 + j * 16 + nl;
      v16bf br = frag8(wW[2], col * 52 + kt * 16 + half * 8);
      v16bf bi = frag8(wW[3], col * 52 + kt * 16 + half * 8);
      dr[j] = WMMA_BF16(ar, br, dr[j]);
      dr[j] = WMMA_BF16(nai, bi, dr[j]);
      di[j] = WMMA_BF16(ai, br, di[j]);
      di[j] = WMMA_BF16(ar, bi, di[j]);
    }
  }
  // ---- bias + soft-shrink + store
#pragma unroll
  for (int j = 0; j < 3; ++j) {
    int n = nh * 48 + j * 16 + nl;
    float b2r = bia[2][n], b2i = bia[3][n];
#pragma unroll
    for (int r = 0; r < 8; ++r) {
      int m = mt * 16 + half * 8 + r;
      int p = pt * 64 + m;                       // (b,gh,wc) flat
      int wc = p % 11, r2 = p / 11, gh = r2 % 20;
      float fu = (float)(gh < 10 ? gh : gh - 20);
      float kk = sqrtf(fu * fu + (float)(wc * wc));
      float q = kk * 0.05f;
      float thr = fmaxf(0.01f * expf(-0.5f * q * q), 0.0005f);
      float zr = dr[j][r] + b2r, zi = di[j][r] + b2i;
      float mag = sqrtf(zr * zr + zi * zi + 1e-8f);
      float sh = fmaxf(mag - thr, 0.f) / mag;
      size_t g = (size_t)p * 768 + nb * 96 + n;
      Zr[g] = zr * sh;
      Zi[g] = zi * sh;
    }
  }
}

// ---------------------------------------------------------------------------
// LayerNorm over E=768, one token per 256-thread block
// ---------------------------------------------------------------------------
__global__ __launch_bounds__(256) void ln_kernel(const float* __restrict__ x,
                                                 const float* __restrict__ w,
                                                 const float* __restrict__ b,
                                                 float* __restrict__ out) {
  __shared__ float red[256];
  const int t = blockIdx.x, tid = threadIdx.x;
  const float* row = x + (size_t)t * 768;
  float s = 0.f;
  for (int i = tid; i < 768; i += 256) s += row[i];
  red[tid] = s; __syncthreads();
  for (int o = 128; o > 0; o >>= 1) { if (tid < o) red[tid] += red[tid + o]; __syncthreads(); }
  float mean = red[0] / 768.f; __syncthreads();
  float v = 0.f;
  for (int i = tid; i < 768; i += 256) { float d = row[i] - mean; v += d * d; }
  red[tid] = v; __syncthreads();
  for (int o = 128; o > 0; o >>= 1) { if (tid < o) red[tid] += red[tid + o]; __syncthreads(); }
  float inv = rsqrtf(red[0] / 768.f + 1e-6f);
  for (int i = tid; i < 768; i += 256)
    out[(size_t)t * 768 + i] = (row[i] - mean) * inv * w[i] + b[i];
}

// ---------------------------------------------------------------------------
// Separable 20-point DFTs (direct; tiny vs GEMM cost). ortho scale 0.05 split
// between forward-W and inverse-W stages.
// ---------------------------------------------------------------------------
__global__ __launch_bounds__(256) void fft_fwd_w(const float* __restrict__ h,
                                                 float* __restrict__ or_,
                                                 float* __restrict__ oi_) {
  size_t idx = (size_t)blockIdx.x * 256 + threadIdx.x;  // (b,gh,wc,e)
  int e = idx % 768; size_t r = idx / 768;
  int wc = r % 11; size_t r2 = r / 11; int gh = r2 % 20, b = r2 / 20;
  const float* base = h + ((size_t)(b * 20 + gh) * 20) * 768 + e;
  float ar = 0.f, ai = 0.f;
  for (int gw = 0; gw < 20; ++gw) {
    int ph = (gw * wc) % 20;
    float sn, cs; __sincosf(-0.31415927f * (float)ph, &sn, &cs);
    float v = base[gw * 768];
    ar += v * cs; ai += v * sn;
  }
  or_[idx] = ar * 0.05f; oi_[idx] = ai * 0.05f;
}

__global__ __launch_bounds__(256) void fft_fwd_h(const float* __restrict__ ir_,
                                                 const float* __restrict__ ii_,
                                                 float* __restrict__ or_,
                                                 float* __restrict__ oi_) {
  size_t idx = (size_t)blockIdx.x * 256 + threadIdx.x;  // (b,k,wc,e)
  int e = idx % 768; size_t r = idx / 768;
  int wc = r % 11; size_t r2 = r / 11; int k = r2 % 20, b = r2 / 20;
  float ar = 0.f, ai = 0.f;
  for (int hh = 0; hh < 20; ++hh) {
    int ph = (hh * k) % 20;
    float sn, cs; __sincosf(-0.31415927f * (float)ph, &sn, &cs);
    size_t g = ((size_t)(b * 20 + hh) * 11 + wc) * 768 + e;
    float xr = ir_[g], xi = ii_[g];
    ar += xr * cs - xi * sn; ai += xr * sn + xi * cs;
  }
  or_[idx] = ar; oi_[idx] = ai;
}

__global__ __launch_bounds__(256) void fft_inv_h(const float* __restrict__ ir_,
                                                 const float* __restrict__ ii_,
                                                 float* __restrict__ or_,
                                                 float* __restrict__ oi_) {
  size_t idx = (size_t)blockIdx.x * 256 + threadIdx.x;  // (b,h,wc,e)
  int e = idx % 768; size_t r = idx / 768;
  int wc = r % 11; size_t r2 = r / 11; int hh = r2 % 20, b = r2 / 20;
  float ar = 0.f, ai = 0.f;
  for (int k = 0; k < 20; ++k) {
    int ph = (hh * k) % 20;
    float sn, cs; __sincosf(0.31415927f * (float)ph, &sn, &cs);
    size_t g = ((size_t)(b * 20 + k) * 11 + wc) * 768 + e;
    float xr = ir_[g], xi = ii_[g];
    ar += xr * cs - xi * sn; ai += xr * sn + xi * cs;
  }
  or_[idx] = ar; oi_[idx] = ai;
}

// inverse rfft along W + residual: t = t + h + z  (0.05 = inverse ortho scale)
__global__ __launch_bounds__(256) void fft_inv_w_res(
    const float* __restrict__ gr, const float* __restrict__ gi,
    const float* __restrict__ h, float* __restrict__ t) {
  size_t idx = (size_t)blockIdx.x * 256 + threadIdx.x;  // (b,gh,gw,e)
  int e = idx % 768; size_t r = idx / 768;
  int gw = r % 20; size_t r2 = r / 20; int gh = r2 % 20, b = r2 / 20;
  float z = 0.f;
  for (int wc = 0; wc < 11; ++wc) {
    float f = (wc == 0 || wc == 10) ? 1.f : 2.f;
    int ph = (gw * wc) % 20;
    float sn, cs; __sincosf(0.31415927f * (float)ph, &sn, &cs);
    size_t g = ((size_t)(b * 20 + gh) * 11 + wc) * 768 + e;
    z += f * (gr[g] * cs - gi[g] * sn);
  }
  t[idx] = t[idx] + h[idx] + 0.05f * z;
}

// im2col for the 8x8 patch embed
__global__ __launch_bounds__(256) void im2col_kernel(const float* __restrict__ x,
                                                     float* __restrict__ ap) {
  size_t idx = (size_t)blockIdx.x * 256 + threadIdx.x;  // m*64+kk
  int kk = idx % 64; size_t m = idx / 64;
  int b = m / 400, tok = m % 400, gh = tok / 20, gw = tok % 20;
  int i = kk / 8, j = kk % 8;
  ap[idx] = x[(size_t)b * 25600 + (size_t)(gh * 8 + i) * 160 + (gw * 8 + j)];
}

// ---------------------------------------------------------------------------
extern "C" void kernel_launch(void* const* d_in, const int* in_sizes, int n_in,
                              void* d_out, int out_size, void* d_ws,
                              size_t ws_size, hipStream_t stream) {
  const float* x       = (const float*)d_in[0];
  const float* patch_w = (const float*)d_in[1];
  const float* patch_b = (const float*)d_in[2];
  const float* pos     = (const float*)d_in[3];
  const float* n1w     = (const float*)d_in[4];
  const float* n1b     = (const float*)d_in[5];
  const float* fw1     = (const float*)d_in[6];
  const float* fb1     = (const float*)d_in[7];
  const float* fw2     = (const float*)d_in[8];
  const float* fb2     = (const float*)d_in[9];
  const float* n2w     = (const float*)d_in[10];
  const float* n2b     = (const float*)d_in[11];
  const float* fc1w    = (const float*)d_in[12];
  const float* fc1b    = (const float*)d_in[13];
  const float* fc2w    = (const float*)d_in[14];
  const float* fc2b    = (const float*)d_in[15];
  const float* normw   = (const float*)d_in[16];
  const float* normb   = (const float*)d_in[17];
  const float* headw   = (const float*)d_in[18];
  const float* headb   = (const float*)d_in[19];
  float* out = (float*)d_out;
  float* ws  = (float*)d_ws;

  // workspace layout (floats); total ~40.3M floats (~161 MB)
  float* t  = ws;                  // 6400*768
  float* h  = ws + 4915200;        // 6400*768
  float* bA = ws + 9830400;        // 4x freq buffers, 16*20*11*768 each
  float* bB = bA + 2703360;
  float* bC = bB + 2703360;
  float* bD = bC + 2703360;
  float* arena  = bD + 2703360;    // max(hidden 6400*3072)
  float* apack  = arena;           // 6400*64 (dead before hidden is live)
  float* hidden = arena;

  // patch embed: im2col + GEMM(6400x768x64) + bias + pos_embed
  im2col_kernel<<<1600, 256, 0, stream>>>(x, apack);
  gemm_bf16<1><<<dim3(6, 50), 256, 0, stream>>>(apack, patch_w, patch_b, pos, t,
                                                6400, 768, 64);

  for (int d = 0; d < 12; ++d) {
    ln_kernel<<<6400, 256, 0, stream>>>(t, n1w + d * 768, n1b + d * 768, h);
    fft_fwd_w<<<10560, 256, 0, stream>>>(h, bA, bB);
    fft_fwd_h<<<10560, 256, 0, stream>>>(bA, bB, bC, bD);
    afno_kernel<<<dim3(55, 8), 256, 0, stream>>>(
        bC, bD, fw1 + (size_t)d * 147456, fb1 + (size_t)d * 1536,
        fw2 + (size_t)d * 147456, fb2 + (size_t)d * 1536, bA, bB);
    fft_inv_h<<<10560, 256, 0, stream>>>(bA, bB, bC, bD);
    fft_inv_w_res<<<19200, 256, 0, stream>>>(bC, bD, h, t);
    // MLP branch
    ln_kernel<<<6400, 256, 0, stream>>>(t, n2w + d * 768, n2b + d * 768, h);
    gemm_bf16<2><<<dim3(24, 50), 256, 0, stream>>>(
        h, fc1w + (size_t)d * 2359296, fc1b + (size_t)d * 3072, nullptr, hidden,
        6400, 3072, 768);
    gemm_bf16<3><<<dim3(6, 50), 256, 0, stream>>>(
        hidden, fc2w + (size_t)d * 2359296, fc2b + (size_t)d * 768, t, t,
        6400, 768, 3072);
  }

  // final norm + head (scatter epilogue writes NCHW output directly)
  ln_kernel<<<6400, 256, 0, stream>>>(t, normw, normb, h);
  gemm_bf16<4><<<dim3(35, 50), 256, 0, stream>>>(h, headw, headb, nullptr, out,
                                                 6400, 4416, 768);
}
